// LongShortAttention_65695819760248
// MI455X (gfx1250) — compile-verified, matrix operand order checked by reading, not run on hardware
//
#include <hip/hip_runtime.h>

typedef __attribute__((ext_vector_type(16))) __bf16 v16bf;
typedef __attribute__((ext_vector_type(8)))  float  v8f;
typedef int v4i_ __attribute__((vector_size(16)));

#define HW    4096
#define W64   64
#define NB    4
#define NHD   2
#define DATT  128
#define CQK   256     /* NH*DATT */
#define HIDD  512
#define EXPD  1024
#define DVU   512
#define WSZ   15
#define WS2   225
#define MAXD  7

#define AS1 __attribute__((address_space(1)))
#define AS3 __attribute__((address_space(3)))

/* ---------------- helpers ---------------- */

static __device__ __forceinline__ int kpair_base(int t, int hi) {
  int b = (t < 4) ? (2 * t) : (16 + 2 * (t - 4));
  return b + (hi ? 8 : 0);
}

static __device__ __forceinline__ v8f zf() {
  v8f c = {0.f, 0.f, 0.f, 0.f, 0.f, 0.f, 0.f, 0.f};
  return c;
}

static __device__ __forceinline__ v16bf zbf() {
  union { unsigned u[8]; v16bf v; } r;
#pragma unroll
  for (int t = 0; t < 8; ++t) r.u[t] = 0u;
  return r.v;
}

/* operand from bf16 buffer row with contiguous K pairs (works for A rows and
   B columns stored (pixel, channel)); 8 dword loads per 32-K chunk. */
static __device__ __forceinline__ v16bf op_from_pc(const __bf16* rowp, int kbase, int hi) {
  union { unsigned u[8]; v16bf v; } r;
#pragma unroll
  for (int t = 0; t < 8; ++t) {
    int k = kbase + kpair_base(t, hi);
    r.u[t] = *(const unsigned*)(rowp + k);
  }
  return r.v;
}

static __device__ __forceinline__ v8f wmma_bf16(v16bf a, v16bf b, v8f c) {
  return __builtin_amdgcn_wmma_f32_16x16x32_bf16(false, a, false, b, (short)0, c, false, false);
}

static __device__ __forceinline__ void store_bf16x2(__bf16* dst, float a, float b) {
  union { __bf16 h[2]; unsigned u; } r;
  r.h[0] = (__bf16)a; r.h[1] = (__bf16)b;
  *(unsigned*)dst = r.u;
}

/* async 16B global -> LDS copy (ASYNCcnt-tracked) */
static __device__ __forceinline__ void async_g2l_b128(const void* g, void* l) {
  __builtin_amdgcn_global_load_async_to_lds_b128(
      (AS1 v4i_*)g, (AS3 v4i_*)l, 0, 0);
}
static __device__ __forceinline__ void wait_async0() {
  asm volatile("s_wait_asynccnt 0" ::: "memory");
}

/* ---------------- 0a) fp32 -> bf16 weight conversion ---------------- */

__global__ __launch_bounds__(256) void k_cvt(const float* __restrict__ src,
                                             __bf16* __restrict__ dst, int n) {
  int i = blockIdx.x * 256 + threadIdx.x;
  if (i < n) dst[i] = (__bf16)src[i];
}

/* ---------------- 0b) (b,C,hw) fp32 -> (b,hw,C) bf16 transpose ---------------- */

__global__ __launch_bounds__(256) void k_tr(const float* __restrict__ src,
                                            __bf16* __restrict__ dst, int C) {
  __shared__ __bf16 tile[32][66];
  int tid = threadIdx.x;
  int p0 = blockIdx.x * 64;
  int c0 = blockIdx.y * 32;
  int b  = blockIdx.z;
#pragma unroll
  for (int s = 0; s < 8; ++s) {
    int idx = tid + s * 256;
    int cl = idx >> 6, pl = idx & 63;
    tile[cl][pl] = (__bf16)src[((size_t)(b * C + c0 + cl)) * HW + p0 + pl];
  }
  __syncthreads();
#pragma unroll
  for (int s = 0; s < 8; ++s) {
    int idx = tid + s * 256;
    int pl = idx >> 5, cl = idx & 31;
    dst[((size_t)(b * HW + p0 + pl)) * C + c0 + cl] = tile[cl][pl];
  }
}

/* ---------------- 1) qk projection: qs16 (scaled) + ks16 ---------------- */

__global__ __launch_bounds__(256) void k_gemm_qk(const __bf16* __restrict__ q16,
                                                 const __bf16* __restrict__ wqk16,
                                                 const float* __restrict__ b_qk,
                                                 __bf16* __restrict__ qs16,
                                                 __bf16* __restrict__ ks16) {
  int tid = threadIdx.x, w = tid >> 5, lane = tid & 31, hi = lane >> 4, lm = lane & 15;
  int m0 = w * 32;                 /* 8 waves x 32 rows = 256 = M */
  int n0 = blockIdx.x * 32;
  int b  = blockIdx.z;
  const __bf16* wr0 = wqk16 + (size_t)(m0 + lm) * DATT;
  const __bf16* wr1 = wr0 + 16 * DATT;
  int p0 = n0 + lm, p1 = p0 + 16;
  const __bf16* r0 = q16 + (size_t)(b * HW + p0) * DATT;
  const __bf16* r1 = q16 + (size_t)(b * HW + p1) * DATT;
  v8f c00 = zf(), c01 = zf(), c10 = zf(), c11 = zf();
#pragma unroll
  for (int cc = 0; cc < 4; ++cc) {
    int kb = cc * 32;
    v16bf a0 = op_from_pc(wr0, kb, hi);
    v16bf a1 = op_from_pc(wr1, kb, hi);
    v16bf b0 = op_from_pc(r0, kb, hi);
    v16bf b1 = op_from_pc(r1, kb, hi);
    c00 = wmma_bf16(a0, b0, c00); c01 = wmma_bf16(a0, b1, c01);
    c10 = wmma_bf16(a1, b0, c10); c11 = wmma_bf16(a1, b1, c11);
  }
  const float qsc = 0.08838834764831845f; /* 1/sqrt(128) */
  size_t row0 = (size_t)(b * HW + p0) * CQK;
  size_t row1 = (size_t)(b * HW + p1) * CQK;
#pragma unroll
  for (int r = 0; r < 8; r += 2) {
    int oa = m0 + r + hi * 8;
    int ob = oa + 16;
    float ba0 = b_qk[oa], ba1 = b_qk[oa + 1];
    float bb0 = b_qk[ob], bb1 = b_qk[ob + 1];
    float v;
    float u;
    v = c00[r] + ba0; u = c00[r + 1] + ba1;
    store_bf16x2(&ks16[row0 + oa], v, u); store_bf16x2(&qs16[row0 + oa], v * qsc, u * qsc);
    v = c01[r] + ba0; u = c01[r + 1] + ba1;
    store_bf16x2(&ks16[row1 + oa], v, u); store_bf16x2(&qs16[row1 + oa], v * qsc, u * qsc);
    v = c10[r] + bb0; u = c10[r + 1] + bb1;
    store_bf16x2(&ks16[row0 + ob], v, u); store_bf16x2(&qs16[row0 + ob], v * qsc, u * qsc);
    v = c11[r] + bb0; u = c11[r + 1] + bb1;
    store_bf16x2(&ks16[row1 + ob], v, u); store_bf16x2(&qs16[row1 + ob], v * qsc, u * qsc);
  }
}

/* ---------------- 2) grouped 1x1 + SiLU for v and u ---------------- */

static __device__ __forceinline__ float silu(float s) {
  return s / (1.f + __expf(-s));
}

__global__ __launch_bounds__(256) void k_gemm_vu(const __bf16* __restrict__ vt16,
                                                 const __bf16* __restrict__ ut16,
                                                 const __bf16* __restrict__ wv16,
                                                 const float* __restrict__ b_v,
                                                 const __bf16* __restrict__ wu16,
                                                 const float* __restrict__ b_u,
                                                 __bf16* __restrict__ vl16,
                                                 float* __restrict__ ulf) {
  int tid = threadIdx.x, w = tid >> 5, lane = tid & 31, hi = lane >> 4, lm = lane & 15;
  int z = blockIdx.z;
  int b = z >> 2, t = (z >> 1) & 1, g = z & 1;
  const __bf16* Wm = t ? wu16 : wv16;
  const float*  Bv = t ? b_u : b_v;
  const __bf16* X  = t ? ut16 : vt16;
  int m0 = (blockIdx.y * 8 + w) * 32;   /* grid.y=2 -> M=512 per group */
  int n0 = blockIdx.x * 32;
  const __bf16* wr0 = Wm + (size_t)(g * 512 + m0 + lm) * 256;
  const __bf16* wr1 = wr0 + 16 * 256;
  int p0 = n0 + lm, p1 = p0 + 16;
  const __bf16* r0 = X + (size_t)(b * HW + p0) * DVU + g * 256;
  const __bf16* r1 = X + (size_t)(b * HW + p1) * DVU + g * 256;
  v8f c00 = zf(), c01 = zf(), c10 = zf(), c11 = zf();
#pragma unroll
  for (int cc = 0; cc < 8; ++cc) {
    int kb = cc * 32;
    v16bf a0 = op_from_pc(wr0, kb, hi);
    v16bf a1 = op_from_pc(wr1, kb, hi);
    v16bf b0 = op_from_pc(r0, kb, hi);
    v16bf b1 = op_from_pc(r1, kb, hi);
    c00 = wmma_bf16(a0, b0, c00); c01 = wmma_bf16(a0, b1, c01);
    c10 = wmma_bf16(a1, b0, c10); c11 = wmma_bf16(a1, b1, c11);
  }
#pragma unroll
  for (int r = 0; r < 8; r += 2) {
    int oa = m0 + r + hi * 8;
    int ob = oa + 16;
    int ga = g * 512 + oa, gb = g * 512 + ob;
    float ba0 = Bv[ga], ba1 = Bv[ga + 1];
    float bb0 = Bv[gb], bb1 = Bv[gb + 1];
    float s00 = silu(c00[r] + ba0), s01 = silu(c00[r + 1] + ba1);
    float s10 = silu(c01[r] + ba0), s11 = silu(c01[r + 1] + ba1);
    float s20 = silu(c10[r] + bb0), s21 = silu(c10[r + 1] + bb1);
    float s30 = silu(c11[r] + bb0), s31 = silu(c11[r + 1] + bb1);
    if (t == 0) {
      store_bf16x2(&vl16[((size_t)(b * HW + p0)) * EXPD + ga], s00, s01);
      store_bf16x2(&vl16[((size_t)(b * HW + p1)) * EXPD + ga], s10, s11);
      store_bf16x2(&vl16[((size_t)(b * HW + p0)) * EXPD + gb], s20, s21);
      store_bf16x2(&vl16[((size_t)(b * HW + p1)) * EXPD + gb], s30, s31);
    } else {
      ulf[((size_t)(b * EXPD + ga)) * HW + p0] = s00;
      ulf[((size_t)(b * EXPD + ga + 1)) * HW + p0] = s01;
      ulf[((size_t)(b * EXPD + ga)) * HW + p1] = s10;
      ulf[((size_t)(b * EXPD + ga + 1)) * HW + p1] = s11;
      ulf[((size_t)(b * EXPD + gb)) * HW + p0] = s20;
      ulf[((size_t)(b * EXPD + gb + 1)) * HW + p0] = s21;
      ulf[((size_t)(b * EXPD + gb)) * HW + p1] = s30;
      ulf[((size_t)(b * EXPD + gb + 1)) * HW + p1] = s31;
    }
  }
}

/* ---------------- 3) relative-position bias GEMM ---------------- */

__global__ __launch_bounds__(256) void k_gemm_rel(const __bf16* __restrict__ qs16,
                                                  const __bf16* __restrict__ wrel16,
                                                  const float* __restrict__ b_rel,
                                                  float* __restrict__ rel) {
  int tid = threadIdx.x, w = tid >> 5, lane = tid & 31, hi = lane >> 4, lm = lane & 15;
  int z = blockIdx.z;
  int b = z >> 1, hd = z & 1;
  int m0 = w * 32;                 /* 8 waves x 32 = 256 rows (225 used) */
  int n0 = blockIdx.x * 32;
  int ma = m0 + lm;       int mac = (ma < WS2) ? ma : (WS2 - 1);
  int mb = m0 + 16 + lm;  int mbc = (mb < WS2) ? mb : (WS2 - 1);
  const __bf16* wr0 = wrel16 + (size_t)(hd * WS2 + mac) * DATT;
  const __bf16* wr1 = wrel16 + (size_t)(hd * WS2 + mbc) * DATT;
  int p0 = n0 + lm, p1 = p0 + 16;
  const __bf16* r0 = qs16 + ((size_t)(b * HW + p0)) * CQK + hd * DATT;
  const __bf16* r1 = qs16 + ((size_t)(b * HW + p1)) * CQK + hd * DATT;
  v8f c00 = zf(), c01 = zf(), c10 = zf(), c11 = zf();
#pragma unroll
  for (int cc = 0; cc < 4; ++cc) {
    int kb = cc * 32;
    v16bf a0 = op_from_pc(wr0, kb, hi);
    v16bf a1 = op_from_pc(wr1, kb, hi);
    v16bf b0 = op_from_pc(r0, kb, hi);
    v16bf b1 = op_from_pc(r1, kb, hi);
    c00 = wmma_bf16(a0, b0, c00); c01 = wmma_bf16(a0, b1, c01);
    c10 = wmma_bf16(a1, b0, c10); c11 = wmma_bf16(a1, b1, c11);
  }
  size_t base = ((size_t)(b * NHD + hd) * WS2) * HW;
#pragma unroll
  for (int r = 0; r < 8; ++r) {
    int oa = m0 + r + hi * 8;
    int ob = oa + 16;
    if (oa < WS2) {
      float bias = b_rel[hd * WS2 + oa];
      rel[base + (size_t)oa * HW + p0] = c00[r] + bias;
      rel[base + (size_t)oa * HW + p1] = c01[r] + bias;
    }
    if (ob < WS2) {
      float bias = b_rel[hd * WS2 + ob];
      rel[base + (size_t)ob * HW + p0] = c10[r] + bias;
      rel[base + (size_t)ob * HW + p1] = c11[r] + bias;
    }
  }
}

/* ---------------- 4) windowed QK^T + bias + mask + softmax -> attn ---------------- */

__global__ __launch_bounds__(128) void k_corr(const __bf16* __restrict__ qs16,
                                              const __bf16* __restrict__ ks16,
                                              const float* __restrict__ rel,
                                              float* __restrict__ attn) {
  __shared__ float lg[16][240];
  __shared__ __attribute__((aligned(16))) float relS[WS2][16];
  int tid = threadIdx.x, w = tid >> 5, lane = tid & 31, hi = lane >> 4, lm = lane & 15;
  int bx = blockIdx.x;
  int y  = bx >> 2;
  int x0 = (bx & 3) << 4;
  int hd = blockIdx.y;
  int b  = blockIdx.z;

  for (int idx = tid; idx < 16 * WS2; idx += 128)
    lg[idx / WS2][idx % WS2] = -1e30f;

  /* async-stage the 225x16 rel tile into LDS */
  const float* relb = rel + ((size_t)(b * NHD + hd) * WS2) * HW + (size_t)(y * W64 + x0);
  for (int idx = tid; idx < WS2 * 4; idx += 128) {
    int o = idx >> 2, qd = idx & 3;
    async_g2l_b128(relb + (size_t)o * HW + qd * 4, &relS[o][qd * 4]);
  }

  int qpix = y * W64 + x0 + lm;
  const __bf16* qrow = qs16 + ((size_t)(b * HW + qpix)) * CQK + hd * DATT;
  v16bf a0 = op_from_pc(qrow,  0, hi);
  v16bf a1 = op_from_pc(qrow, 32, hi);
  v16bf a2 = op_from_pc(qrow, 64, hi);
  v16bf a3 = op_from_pc(qrow, 96, hi);

  wait_async0();
  __syncthreads();

  for (int s = w; s < 30; s += 4) {
    int d = s >> 1, t = s & 1;
    int yk = y + d - MAXD;
    if (yk < 0 || yk >= W64) continue;
    int xk = x0 - MAXD + t * 16 + lm;
    bool xv = (xk >= 0 && xk < W64);
    int xkc = xv ? xk : 0;
    const __bf16* krow = ks16 + ((size_t)(b * HW + yk * W64 + xkc)) * CQK + hd * DATT;
    v8f c = zf();
    v16bf bb;
    bb = op_from_pc(krow,  0, hi); if (!xv) bb = zbf(); c = wmma_bf16(a0, bb, c);
    bb = op_from_pc(krow, 32, hi); if (!xv) bb = zbf(); c = wmma_bf16(a1, bb, c);
    bb = op_from_pc(krow, 64, hi); if (!xv) bb = zbf(); c = wmma_bf16(a2, bb, c);
    bb = op_from_pc(krow, 96, hi); if (!xv) bb = zbf(); c = wmma_bf16(a3, bb, c);
#pragma unroll
    for (int r = 0; r < 8; ++r) {
      int i  = r + hi * 8;
      int o7 = t * 16 + lm - i;   /* dx + 7 */
      if (o7 >= 0 && o7 < WSZ && xv) {
        int o = d * WSZ + o7;
        lg[i][o] = c[r] + relS[o][i];
      }
    }
  }
  __syncthreads();

  /* softmax over 225, 8 lanes per pixel */
  int pix = tid >> 3, g = tid & 7;
  int pimg = y * W64 + x0 + pix;
  float mx = -1e30f;
  for (int o = g; o < WS2; o += 8) mx = fmaxf(mx, lg[pix][o]);
  mx = fmaxf(mx, __shfl_xor(mx, 1, 8));
  mx = fmaxf(mx, __shfl_xor(mx, 2, 8));
  mx = fmaxf(mx, __shfl_xor(mx, 4, 8));
  float sm = 0.f;
  for (int o = g; o < WS2; o += 8) sm += __expf(lg[pix][o] - mx);
  sm += __shfl_xor(sm, 1, 8);
  sm += __shfl_xor(sm, 2, 8);
  sm += __shfl_xor(sm, 4, 8);
  float inv = 1.f / sm;
  float* ab = attn + ((size_t)(b * NHD + hd) * WS2) * HW + pimg;
  for (int o = g; o < WS2; o += 8)
    ab[(size_t)o * HW] = __expf(lg[pix][o] - mx) * inv;
}

/* ---------------- 5) attn x V (WMMA) fused with u-gate -> x ---------------- */

__global__ __launch_bounds__(256) void k_agg(const float* __restrict__ attn,
                                             const __bf16* __restrict__ vl16,
                                             const float* __restrict__ ulf,
                                             float* __restrict__ xbuf) {
  __shared__ __bf16 As[16][32];
  __shared__ __attribute__((aligned(16))) float attS[WSZ][16];
  int tid = threadIdx.x, w = tid >> 5, lane = tid & 31, hi = lane >> 4, lm = lane & 15;
  int bx = blockIdx.x;
  int y  = bx >> 2;
  int x0 = (bx & 3) << 4;
  int hd = blockIdx.y;
  int b  = blockIdx.z;

  const float* ab = attn + ((size_t)(b * NHD + hd) * WS2) * HW + (size_t)(y * W64 + x0);
  v8f acc[4];
#pragma unroll
  for (int qd = 0; qd < 4; ++qd) acc[qd] = zf();

  for (int d = 0; d < WSZ; ++d) {
    int yk = y + d - MAXD;
    bool yv = (yk >= 0 && yk < W64);
    __syncthreads();
    if (yv && tid < WSZ * 4) {
      int o = tid >> 2, qd = tid & 3;
      async_g2l_b128(ab + (size_t)(d * WSZ + o) * HW + qd * 4, &attS[o][qd * 4]);
    }
    wait_async0();
    __syncthreads();
    if (yv) {
      for (int idx = tid; idx < 512; idx += 256) {
        int i = idx >> 5, j = idx & 31;
        int dj = j - i;
        As[i][j] = (dj >= 0 && dj < WSZ) ? (__bf16)attS[dj][i] : (__bf16)0.f;
      }
    }
    __syncthreads();
    if (!yv) continue;

    v16bf a;
    {
      union { unsigned u[8]; v16bf v; } r;
#pragma unroll
      for (int t2 = 0; t2 < 8; ++t2) {
        int k = kpair_base(t2, hi);
        r.u[t2] = *(const unsigned*)&As[lm][k];
      }
      a = r.v;
    }
#pragma unroll
    for (int qd = 0; qd < 4; ++qd) {
      int cb = (w + 8 * qd) * 16;
      int cp = cb + lm;
      int vc = ((cp >> 8) * 512) + hd * 256 + (cp & 255);
      const __bf16* vb = vl16 + ((size_t)(b * HW + yk * W64)) * EXPD + vc;
      union { __bf16 h[16]; v16bf v; } rb;
#pragma unroll
      for (int t2 = 0; t2 < 8; ++t2) {
        int k  = kpair_base(t2, hi);
        int xa = x0 - MAXD + k;
        int xc = x0 - MAXD + k + 1;
        rb.h[2 * t2 + 0] = (xa >= 0 && xa < W64) ? vb[(size_t)xa * EXPD] : (__bf16)0.f;
        rb.h[2 * t2 + 1] = (xc >= 0 && xc < W64) ? vb[(size_t)xc * EXPD] : (__bf16)0.f;
      }
      acc[qd] = wmma_bf16(a, rb.v, acc[qd]);
    }
  }

#pragma unroll
  for (int qd = 0; qd < 4; ++qd) {
    int cb = (w + 8 * qd) * 16;
    int cp = cb + lm;
    int e  = hd * HIDD + cp;
    int uc = ((cp >> 8) * 512) + hd * 256 + (cp & 255);
#pragma unroll
    for (int r = 0; r < 8; ++r) {
      int i = r + hi * 8;
      int p = y * W64 + x0 + i;
      float uval = ulf[((size_t)(b * EXPD + uc)) * HW + p];
      xbuf[((size_t)(b * EXPD + e)) * HW + p] = acc[qd][r] * uval;
    }
  }
}

/* ---------------- 6) depthwise 3x3 -> y16 (pixel, channel) bf16 ---------------- */

__global__ __launch_bounds__(256) void k_dw(const float* __restrict__ xbuf,
                                            const float* __restrict__ w_dw,
                                            const float* __restrict__ b_dw,
                                            __bf16* __restrict__ y16) {
  int e   = blockIdx.y;
  int pid = blockIdx.x * 256 + threadIdx.x;
  int b   = pid >> 12;
  int p   = pid & 4095;
  int yy  = p >> 6, xx = p & 63;
  const float* wd = w_dw + (size_t)e * 9;
  const float* xb = xbuf + ((size_t)(b * EXPD + e)) * HW;
  float s = b_dw[e];
#pragma unroll
  for (int ky = 0; ky < 3; ++ky) {
    int y2 = yy + ky - 1;
    if (y2 < 0 || y2 >= W64) continue;
#pragma unroll
    for (int kx = 0; kx < 3; ++kx) {
      int x2 = xx + kx - 1;
      if (x2 < 0 || x2 >= W64) continue;
      s += wd[ky * 3 + kx] * xb[y2 * W64 + x2];
    }
  }
  y16[((size_t)(b * HW + p)) * EXPD + e] = (__bf16)s;
}

/* ---------------- 7) output projection GEMM ---------------- */

__global__ __launch_bounds__(256) void k_proj(const __bf16* __restrict__ y16,
                                              const __bf16* __restrict__ wproj16,
                                              const float* __restrict__ b_proj,
                                              float* __restrict__ out) {
  int tid = threadIdx.x, w = tid >> 5, lane = tid & 31, hi = lane >> 4, lm = lane & 15;
  int m0 = (blockIdx.y * 8 + w) * 32;   /* grid.y=2 -> M=512 */
  int n0 = blockIdx.x * 32;
  int b  = blockIdx.z;
  const __bf16* wr0 = wproj16 + (size_t)(m0 + lm) * EXPD;
  const __bf16* wr1 = wr0 + 16 * EXPD;
  int p0 = n0 + lm, p1 = p0 + 16;
  const __bf16* r0 = y16 + ((size_t)(b * HW + p0)) * EXPD;
  const __bf16* r1 = y16 + ((size_t)(b * HW + p1)) * EXPD;
  v8f c00 = zf(), c01 = zf(), c10 = zf(), c11 = zf();
  for (int cc = 0; cc < 32; ++cc) {
    int kb = cc * 32;
    if (cc + 4 < 32) {
      __builtin_prefetch(r0 + kb + 128, 0, 1);
      __builtin_prefetch(r1 + kb + 128, 0, 1);
    }
    v16bf a0 = op_from_pc(wr0, kb, hi);
    v16bf a1 = op_from_pc(wr1, kb, hi);
    v16bf b0 = op_from_pc(r0, kb, hi);
    v16bf b1 = op_from_pc(r1, kb, hi);
    c00 = wmma_bf16(a0, b0, c00); c01 = wmma_bf16(a0, b1, c01);
    c10 = wmma_bf16(a1, b0, c10); c11 = wmma_bf16(a1, b1, c11);
  }
#pragma unroll
  for (int r = 0; r < 8; r += 2) {
    int oa = m0 + r + hi * 8;
    int ob = oa + 16;
    float2 vv;
    vv.x = c00[r] + b_proj[oa]; vv.y = c00[r + 1] + b_proj[oa + 1];
    *(float2*)&out[((size_t)p0 * NB + b) * DVU + oa] = vv;
    vv.x = c01[r] + b_proj[oa]; vv.y = c01[r + 1] + b_proj[oa + 1];
    *(float2*)&out[((size_t)p1 * NB + b) * DVU + oa] = vv;
    vv.x = c10[r] + b_proj[ob]; vv.y = c10[r + 1] + b_proj[ob + 1];
    *(float2*)&out[((size_t)p0 * NB + b) * DVU + ob] = vv;
    vv.x = c11[r] + b_proj[ob]; vv.y = c11[r + 1] + b_proj[ob + 1];
    *(float2*)&out[((size_t)p1 * NB + b) * DVU + ob] = vv;
  }
}

/* ---------------- host launcher ---------------- */

extern "C" void kernel_launch(void* const* d_in, const int* in_sizes, int n_in,
                              void* d_out, int out_size, void* d_ws, size_t ws_size,
                              hipStream_t stream) {
  (void)in_sizes; (void)n_in; (void)out_size; (void)ws_size;
  const float* q      = (const float*)d_in[0];
  /* d_in[1] (k) ignored, matching the reference. */
  const float* vin    = (const float*)d_in[2];
  const float* uin    = (const float*)d_in[3];
  const float* w_qk   = (const float*)d_in[5];
  const float* b_qk   = (const float*)d_in[6];
  const float* w_v    = (const float*)d_in[7];
  const float* b_v    = (const float*)d_in[8];
  const float* w_u    = (const float*)d_in[9];
  const float* b_u    = (const float*)d_in[10];
  const float* w_rel  = (const float*)d_in[11];
  const float* b_rel  = (const float*)d_in[12];
  const float* w_dw   = (const float*)d_in[13];
  const float* b_dw   = (const float*)d_in[14];
  const float* w_proj = (const float*)d_in[15];
  const float* b_proj = (const float*)d_in[16];

  char* ws = (char*)d_ws;
  size_t off = 0;
  __bf16* qs16 = (__bf16*)(ws + off); off += (size_t)NB * HW * CQK * 2;
  __bf16* ks16 = (__bf16*)(ws + off); off += (size_t)NB * HW * CQK * 2;
  __bf16* vl16 = (__bf16*)(ws + off); off += (size_t)NB * HW * EXPD * 2;
  float*  ulf  = (float*)(ws + off);  off += (size_t)NB * EXPD * HW * 4;
  float*  rel  = (float*)(ws + off);  off += (size_t)NB * NHD * WS2 * HW * 4;
  float*  xbuf = (float*)(ws + off);  off += (size_t)NB * EXPD * HW * 4;
  __bf16* y16  = (__bf16*)(ws + off); off += (size_t)NB * HW * EXPD * 2;
  __bf16* q16  = (__bf16*)(ws + off); off += (size_t)NB * HW * DATT * 2;
  __bf16* vt16 = (__bf16*)(ws + off); off += (size_t)NB * HW * DVU * 2;
  __bf16* ut16 = (__bf16*)(ws + off); off += (size_t)NB * HW * DVU * 2;
  __bf16* wqk16   = (__bf16*)(ws + off); off += (size_t)CQK * DATT * 2;
  __bf16* wv16    = (__bf16*)(ws + off); off += (size_t)EXPD * 256 * 2;
  __bf16* wu16    = (__bf16*)(ws + off); off += (size_t)EXPD * 256 * 2;
  __bf16* wrel16  = (__bf16*)(ws + off); off += (size_t)NHD * WS2 * DATT * 2;
  __bf16* wproj16 = (__bf16*)(ws + off); off += (size_t)DVU * EXPD * 2;

  float* out  = (float*)d_out;
  float* attn = out + (size_t)HW * NB * DVU;

  /* stage 0: convert weights, transpose+convert activations */
  hipLaunchKernelGGL(k_cvt, dim3(128),  dim3(256), 0, stream, w_qk,   wqk16,   CQK * DATT);
  hipLaunchKernelGGL(k_cvt, dim3(1024), dim3(256), 0, stream, w_v,    wv16,    EXPD * 256);
  hipLaunchKernelGGL(k_cvt, dim3(1024), dim3(256), 0, stream, w_u,    wu16,    EXPD * 256);
  hipLaunchKernelGGL(k_cvt, dim3(225),  dim3(256), 0, stream, w_rel,  wrel16,  NHD * WS2 * DATT);
  hipLaunchKernelGGL(k_cvt, dim3(2048), dim3(256), 0, stream, w_proj, wproj16, DVU * EXPD);
  hipLaunchKernelGGL(k_tr, dim3(64, 4, 4),  dim3(256), 0, stream, q,   q16,  DATT);
  hipLaunchKernelGGL(k_tr, dim3(64, 16, 4), dim3(256), 0, stream, vin, vt16, DVU);
  hipLaunchKernelGGL(k_tr, dim3(64, 16, 4), dim3(256), 0, stream, uin, ut16, DVU);

  /* main pipeline */
  hipLaunchKernelGGL(k_gemm_qk,  dim3(128, 1, 4),   dim3(256), 0, stream,
                     q16, wqk16, b_qk, qs16, ks16);
  hipLaunchKernelGGL(k_gemm_vu,  dim3(128, 2, 16),  dim3(256), 0, stream,
                     vt16, ut16, wv16, b_v, wu16, b_u, vl16, ulf);
  hipLaunchKernelGGL(k_gemm_rel, dim3(128, 1, 8),   dim3(256), 0, stream,
                     qs16, wrel16, b_rel, rel);
  hipLaunchKernelGGL(k_corr,     dim3(256, 2, 4),   dim3(128), 0, stream,
                     qs16, ks16, rel, attn);
  hipLaunchKernelGGL(k_agg,      dim3(256, 2, 4),   dim3(256), 0, stream,
                     attn, vl16, ulf, xbuf);
  hipLaunchKernelGGL(k_dw,       dim3(64, 1024, 1), dim3(256), 0, stream,
                     xbuf, w_dw, b_dw, y16);
  hipLaunchKernelGGL(k_proj,     dim3(128, 2, 4),   dim3(256), 0, stream,
                     y16, wproj16, b_proj, out);
}
